// DGCN_9543417332137
// MI455X (gfx1250) — matrix operand order repaired and checked by reading, not compiled, not run on hardware
//
#include <hip/hip_runtime.h>
#include <hip/hip_bf16.h>

// ---------------------------------------------------------------------------
// DGCN forward for MI455X (gfx1250). Heavy GEMMs via v_wmma_f32_16x16x32_f16,
// B-operand staging via Tensor Data Mover (tensor_load_to_lds) when available.
// ---------------------------------------------------------------------------

typedef __attribute__((ext_vector_type(16))) _Float16 v16h;
typedef __attribute__((ext_vector_type(8)))  _Float16 v8h;
typedef __attribute__((ext_vector_type(8)))  float    v8f;
typedef __attribute__((ext_vector_type(4)))  unsigned int u32x4;
typedef __attribute__((ext_vector_type(8)))  int          i32x8;
typedef __attribute__((ext_vector_type(4)))  int          i32x4;

#define LDSB 72   // padded LDS stride (halves); 72*2=144 bytes, 16B aligned rows

#if defined(__has_builtin)
#if __has_builtin(__builtin_amdgcn_tensor_load_to_lds) && __has_builtin(__builtin_amdgcn_s_wait_tensorcnt)
#define USE_TDM 1
#endif
#endif
#ifndef USE_TDM
#define USE_TDM 0
#endif

// ---------------------------------------------------------------------------
// Generic batched WMMA GEMM: C_bz = act(A_bz[MxK] * B_bz[KxN] + bias)
//   block = 256 threads = 8 waves arranged 2 (rows) x 4 (cols) of 16x16 tiles
//   requires: M%32==0, K%32==0, lda%8==0, ldb%8==0 (N any mult of 16)
// ---------------------------------------------------------------------------
template<bool BIAS, bool RELU, bool ACC, bool OUT16>
__global__ __launch_bounds__(256)
void gemm_wmma(const _Float16* __restrict__ A, const _Float16* __restrict__ B,
               void* __restrict__ Cv, const float* __restrict__ bias,
               int M, int N, int K, int lda, int ldb, int ldc,
               long sA, long sB, long sC)
{
    __shared__ _Float16 Bs[32 * LDSB];

    const int tid  = threadIdx.x;
    const int lane = tid & 31;
    const int wave = tid >> 5;
    const int wrow = wave >> 2;   // 0..1
    const int wcol = wave & 3;    // 0..3

    const long bz = blockIdx.z;
    A += bz * sA;
    B += bz * sB;
    float*    Cf = (float*)Cv     + bz * sC;
    _Float16* Ch = (_Float16*)Cv  + bz * sC;

    const int rowBase = blockIdx.y * 32 + wrow * 16;
    const int colBase = blockIdx.x * 64;
    const int myCol   = colBase + wcol * 16 + (lane & 15);

    // A-fragment addressing (16-bit A 16x32 layout, wave32)
    const int arow = rowBase + (lane & 15);
    const int koff = (lane >> 4) * 8;          // lanes 0-15: K 0..7/16..23 ; 16-31: 8..15/24..31
    // B-fragment addressing from LDS (B 32x16 layout)
    const int bcolL = wcol * 16 + (lane & 15);
    const int bkL   = (lane >> 4) * 8;

    v8f acc;
#pragma unroll
    for (int i = 0; i < 8; ++i) acc[i] = 0.f;

    for (int kk = 0; kk < K; kk += 32) {
#if USE_TDM
        // ---- stage B tile [32 x 64] into LDS via Tensor Data Mover ----
        // D# groups per CDNA5 ISA ch.8: 2D tile, 2-byte elems, LDS row pad of
        // 4 DWORDs (8 halves) every 32 DWORDs (64 halves) -> LDSB==72 stride.
        if (wave == 0) {
            const unsigned lds = (unsigned)(uintptr_t)(void*)Bs;
            const unsigned long long ga =
                (unsigned long long)(uintptr_t)(B + (long)kk * ldb + colBase);
            u32x4 g0;
            g0[0] = 1u;                                            // count=1 user desc
            g0[1] = lds;                                           // lds_addr
            g0[2] = (unsigned)ga;                                  // global_addr[31:0]
            g0[3] = (unsigned)((ga >> 32) & 0x1FFFFFFu) | (2u << 30); // addr[56:32] | type=2
            i32x8 g1;
            g1[0] = (1 << 16) | (1 << 20) | (4 << 22) | (3 << 25); // 2B elems, pad_en, int=32dw, amt=4dw
            g1[1] = (int)(((unsigned)N & 0xffffu) << 16);          // tensor_dim0[15:0]
            g1[2] = (int)((((unsigned)N >> 16) & 0xffffu) | (32u << 16)); // dim0[31:16], tensor_dim1=32
            g1[3] = (int)(64u << 16);                              // tile_dim0 = 64
            g1[4] = 32;                                            // tile_dim1 = 32, tile_dim2 = 0
            g1[5] = ldb;                                           // tensor_dim0_stride[31:0]
            g1[6] = 0;
            g1[7] = 0;
            i32x4 g2; g2[0] = g2[1] = g2[2] = g2[3] = 0;
            i32x4 g3; g3[0] = g3[1] = g3[2] = g3[3] = 0;
            i32x8 g4;
#pragma unroll
            for (int e = 0; e < 8; ++e) g4[e] = 0;
            __builtin_amdgcn_tensor_load_to_lds(g0, g1, g2, g3, g4, 0);
            __builtin_amdgcn_s_wait_tensorcnt(0);                  // s_wait_tensorcnt 0
        }
        __syncthreads();
#else
        // ---- fallback: cooperative vector load of B tile [32 x 64] ----
        {
            const int r  = tid >> 3;            // 0..31
            const int c0 = (tid & 7) * 8;       // 0..56
            const int gc = colBase + c0;
            const _Float16* src = B + (long)(kk + r) * ldb;
            _Float16* dst = &Bs[r * LDSB + c0];
            if (gc + 8 <= N) {
                *(v8h*)dst = *(const v8h*)(src + gc);
            } else {
#pragma unroll
                for (int e = 0; e < 8; ++e) {
                    int c = gc + e;
                    dst[e] = src[c < N ? c : (N - 1)];
                }
            }
        }
        __syncthreads();
#endif

        // ---- A fragment: two 16-byte loads per lane ----
        const _Float16* ap = A + (long)arow * lda + kk + koff;
        v8h lo = *(const v8h*)ap;
        v8h hi = *(const v8h*)(ap + 16);
        v16h af = __builtin_shufflevector(lo, hi, 0, 1, 2, 3, 4, 5, 6, 7,
                                          8, 9, 10, 11, 12, 13, 14, 15);

        // ---- B fragment from LDS (B 32x16 lane layout) ----
        v16h bf;
#pragma unroll
        for (int e = 0; e < 16; ++e) bf[e] = Bs[(bkL + e + (e >> 3) * 8) * LDSB + bcolL];

        acc = __builtin_amdgcn_wmma_f32_16x16x32_f16(false, af, false, bf,
                                                     (short)0, acc, false, false);
        __syncthreads();

        if (kk + 32 < K)   // prefetch next A K-tile -> global_prefetch_b8
            __builtin_prefetch(A + (long)arow * lda + kk + 32, 0, 0);
    }

    // ---- epilogue (C/D layout: vgpr i -> M = (lane>=16 ? 8 : 0)+i, N = lane&15) ----
#pragma unroll
    for (int i = 0; i < 8; ++i) {
        const int row = rowBase + (lane >> 4) * 8 + i;
        const int col = myCol;
        if (col < N) {
            float v = acc[i];
            if (BIAS) v += bias[col];
            if (RELU) v = v > 0.f ? v : 0.f;
            const long idx = (long)row * ldc + col;
            if (ACC)            Cf[idx] += v;
            else if (OUT16)     Ch[idx]  = (_Float16)v;
            else                Cf[idx]  = v;
        }
    }
}

template<bool BIAS, bool RELU, bool ACC, bool OUT16>
static void gemmL(hipStream_t st, const _Float16* A, const _Float16* B, void* C,
                  const float* bias, int M, int N, int K, int lda, int ldb, int ldc,
                  long sA, long sB, long sC, int batch)
{
    dim3 g((unsigned)((N + 63) / 64), (unsigned)(M / 32), (unsigned)batch);
    gemm_wmma<BIAS, RELU, ACC, OUT16><<<g, dim3(256), 0, st>>>(
        A, B, C, bias, M, N, K, lda, ldb, ldc, sA, sB, sC);
}

// ---------------------------------------------------------------------------
// small helper kernels
// ---------------------------------------------------------------------------
__global__ void k_f32_to_f16(const float* __restrict__ s, _Float16* __restrict__ d, int n)
{
    int i = blockIdx.x * 256 + threadIdx.x;
    if (i < n) d[i] = (_Float16)s[i];
}

__global__ void k_zero_f16(_Float16* p, int n)
{
    int i = blockIdx.x * 256 + threadIdx.x;
    if (i < n) p[i] = (_Float16)0.f;
}

// src_seq [B,S,T,F] f32 -> [B,T,S,F] f16
__global__ void k_transpose_src(const float* __restrict__ s, _Float16* __restrict__ d)
{
    int i = blockIdx.x * 256 + threadIdx.x;       // B*S*T*F = 4194304
    if (i >= 8 * 1024 * 16 * 32) return;
    int f  = i & 31;
    int t  = (i >> 5) & 15;
    int ss = (i >> 9) & 1023;
    int b  = i >> 19;
    d[(((long)(b * 16 + t) * 1024 + ss) * 32) + f] = (_Float16)s[i];
}

// GRU gate fusion: gi/gh f32 [8192,192] (torch order r,z,n), h f16 [8192,64]
__global__ void k_gru_gate(const float* __restrict__ gi, const float* __restrict__ gh,
                           _Float16* __restrict__ h, _Float16* __restrict__ rout, int t)
{
    int i = blockIdx.x * 256 + threadIdx.x;       // n*64+d
    if (i >= 8192 * 64) return;
    int d = i & 63, n = i >> 6;
    int b = n >> 10, s = n & 1023;
    const float* gib = gi + (long)n * 192;
    const float* ghb = gh + (long)n * 192;
    float r  = 1.f / (1.f + __expf(-(gib[d]       + ghb[d])));
    float u  = 1.f / (1.f + __expf(-(gib[64 + d]  + ghb[64 + d])));
    float nn = tanhf(gib[128 + d] + r * ghb[128 + d]);
    float hnew = (1.f - u) * nn + u * (float)h[i];
    h[i] = (_Float16)hnew;
    rout[((long)((b * 16 + t) * 1024 + s)) * 64 + d] = (_Float16)hnew;
}

// causal MHA core: per thread (n, head, qrow). q/k/v in [B,T,S,64] f16, head dim 16
__global__ void k_attention(const _Float16* __restrict__ q, const _Float16* __restrict__ k,
                            const _Float16* __restrict__ v, _Float16* __restrict__ o)
{
    int tid = blockIdx.x * 256 + threadIdx.x;     // 8192*4*16
    if (tid >= 8192 * 4 * 16) return;
    int qr = tid & 15, pair = tid >> 4;
    int head = pair & 3, n = pair >> 2;
    int b = n >> 10, s = n & 1023;
    const long TS   = 1024 * 64;
    const long base = ((long)(b * 16) * 1024 + s) * 64 + head * 16;

    float Q[16];
#pragma unroll
    for (int d = 0; d < 16; ++d) Q[d] = (float)q[base + (long)qr * TS + d];

    float sc[16], m = -1e30f;
#pragma unroll
    for (int kt = 0; kt < 16; ++kt) {
        sc[kt] = 0.f;
        if (kt <= qr) {
            float a = 0.f;
#pragma unroll
            for (int d = 0; d < 16; ++d) a += Q[d] * (float)k[base + (long)kt * TS + d];
            a *= 0.25f;                       // 1/sqrt(DK=16)
            sc[kt] = a;
            if (a > m) m = a;
        }
    }
    float p[16], sum = 0.f;
#pragma unroll
    for (int kt = 0; kt < 16; ++kt) {
        p[kt] = (kt <= qr) ? __expf(sc[kt] - m) : 0.f;
        sum += p[kt];
    }
    float acc[16];
#pragma unroll
    for (int d = 0; d < 16; ++d) acc[d] = 0.f;
#pragma unroll
    for (int kt = 0; kt < 16; ++kt) {
        if (kt <= qr) {
            float pk = p[kt];
#pragma unroll
            for (int d = 0; d < 16; ++d) acc[d] += pk * (float)v[base + (long)kt * TS + d];
        }
    }
    const float inv = 1.f / sum;
#pragma unroll
    for (int d = 0; d < 16; ++d)
        o[base + (long)qr * TS + d] = (_Float16)(acc[d] * inv);
}

// residual + layernorm; input rows in [B,T,S,64] order, output in [B,S,T,64] order
__global__ void k_ln(const float* __restrict__ ob, const _Float16* __restrict__ r2,
                     const float* __restrict__ gam, const float* __restrict__ bet,
                     _Float16* __restrict__ out)
{
    int row = blockIdx.x * 256 + threadIdx.x;     // 131072 rows
    if (row >= 8 * 16 * 1024) return;
    int s = row & 1023, t = (row >> 10) & 15, b = row >> 14;
    const float*    x = ob + (long)row * 64;
    const _Float16* r = r2 + (long)row * 64;
    float mu = 0.f;
    for (int d = 0; d < 64; ++d) mu += x[d] + (float)r[d];
    mu *= (1.f / 64.f);
    float var = 0.f;
    for (int d = 0; d < 64; ++d) {
        float val = x[d] + (float)r[d] - mu;
        var += val * val;
    }
    var *= (1.f / 64.f);
    const float inv = rsqrtf(var + 1e-6f);
    _Float16* o = out + (((long)(b * 1024 + s) * 16) + t) * 64;
    for (int d = 0; d < 64; ++d) {
        float val = x[d] + (float)r[d] - mu;
        o[d] = (_Float16)(val * inv * gam[d] + bet[d]);
    }
}

// final projection: out[n,c] = sum_k hid[n,k] * prj_w[k,c]  (N=2)
__global__ void k_proj(const _Float16* __restrict__ hid, const float* __restrict__ w,
                       float* __restrict__ out)
{
    int i = blockIdx.x * 256 + threadIdx.x;       // n*2+c
    if (i >= 8192 * 2) return;
    int c = i & 1, n = i >> 1;
    const _Float16* h = hid + (long)n * 256;
    float a = 0.f;
    for (int k2 = 0; k2 < 256; ++k2) a += (float)h[k2] * w[k2 * 2 + c];
    out[i] = a;
}

// ---------------------------------------------------------------------------
extern "C" void kernel_launch(void* const* d_in, const int* in_sizes, int n_in,
                              void* d_out, int out_size, void* d_ws, size_t ws_size,
                              hipStream_t stream)
{
    (void)in_sizes; (void)n_in; (void)out_size; (void)ws_size;

    // ---- input pointers (setup_inputs order) ----
    const float* src    = (const float*)d_in[0];
    const float* Hf     = (const float*)d_in[1];
    const float* adjf   = (const float*)d_in[2];
    const float* gc1_w  = (const float*)d_in[4];
    const float* gc1_b  = (const float*)d_in[5];
    const float* gc2_w  = (const float*)d_in[6];
    const float* gc2_b  = (const float*)d_in[7];
    const float* m_w    = (const float*)d_in[8];
    const float* m_b    = (const float*)d_in[9];
    const float* lin_w  = (const float*)d_in[10];
    const float* lin_b  = (const float*)d_in[11];
    const float* wih0   = (const float*)d_in[12];
    const float* whh0   = (const float*)d_in[13];
    const float* bih0   = (const float*)d_in[14];
    const float* bhh0   = (const float*)d_in[15];
    const float* wih1   = (const float*)d_in[16];
    const float* whh1   = (const float*)d_in[17];
    const float* bih1   = (const float*)d_in[18];
    const float* bhh1   = (const float*)d_in[19];
    const float* wq     = (const float*)d_in[20];
    const float* wk     = (const float*)d_in[21];
    const float* wv     = (const float*)d_in[22];
    const float* fc_w   = (const float*)d_in[23];
    const float* fc_b   = (const float*)d_in[24];
    const float* ln_g   = (const float*)d_in[25];
    const float* ln_b   = (const float*)d_in[26];
    const float* l_w    = (const float*)d_in[27];
    const float* l_b    = (const float*)d_in[28];
    const float* prj_w  = (const float*)d_in[29];
    float* out = (float*)d_out;

    // ---- workspace layout (deterministic bump allocator + region reuse) ----
    char* ws = (char*)d_ws;
    size_t off = 0;
    auto take = [&](size_t bytes) { size_t r = off; off += (bytes + 255) & ~(size_t)255; return r; };

    const long STOCK = 1024, Bb = 8, Tt = 16, FEAT = 32, NHID = 256, DWV = 64, RNN = 64;
    const long NTOK = Bb * Tt * STOCK;            // 131072

    const size_t off_adjh  = take(STOCK * STOCK * 2);
    const size_t off_Hh    = take(STOCK * 32 * 2);
    const size_t off_mwh   = take(32 * STOCK * 2);
    const size_t off_Hnewh = take(STOCK * STOCK * 2);
    const size_t off_gc1h  = take(FEAT * NHID * 2);
    const size_t off_gc2h  = take(NHID * FEAT * 2);
    const size_t off_linh  = take(FEAT * DWV * 2);
    const size_t off_wih0  = take(DWV * 192 * 2);
    const size_t off_whh0  = take(RNN * 192 * 2);
    const size_t off_wih1  = take(RNN * 192 * 2);
    const size_t off_whh1  = take(RNN * 192 * 2);
    const size_t off_wqh   = take(RNN * 64 * 2);
    const size_t off_wkh   = take(RNN * 64 * 2);
    const size_t off_wvh   = take(RNN * 64 * 2);
    const size_t off_fch   = take(64 * RNN * 2);
    const size_t off_lwh   = take(1024 * NHID * 2);
    const size_t off_xh    = take(NTOK * FEAT * 2);
    const size_t reg1 = take(67108864);  // s1  | later: xlin, r1, r2, o1 (4 x 16MB)
    const size_t reg2 = take(67108864);  // h   | later: q,k,v | obuf(f32), olnb
    const size_t reg3 = take(8388608);   // s2  | later: gi | hid
    const size_t reg4 = take(16777216);  // z   | later: gh, hstate
    const size_t reg5 = take(8388608);   // zh

    const size_t off_s1   = reg1;
    const size_t off_xlin = reg1;
    const size_t off_r1   = reg1 + 16777216;
    const size_t off_r2   = reg1 + 33554432;
    const size_t off_o1   = reg1 + 50331648;
    const size_t off_h    = reg2;
    const size_t off_q    = reg2;
    const size_t off_k    = reg2 + 16777216;
    const size_t off_v    = reg2 + 33554432;
    const size_t off_ob   = reg2;                 // f32, after q/k/v dead
    const size_t off_olnb = reg2 + 33554432;
    const size_t off_s2   = reg3;
    const size_t off_gi   = reg3;
    const size_t off_hid  = reg3;
    const size_t off_z    = reg4;                 // f32
    const size_t off_gh   = reg4;                 // f32, after z dead
    const size_t off_hst  = reg4 + 8388608;
    const size_t off_zh   = reg5;

    auto F16 = [&](size_t o_) { return (_Float16*)(ws + o_); };
    auto F32 = [&](size_t o_) { return (float*)(ws + o_); };

    auto cvt = [&](const float* s, size_t o_, int n) {
        k_f32_to_f16<<<dim3((n + 255) / 256), dim3(256), 0, stream>>>(s, F16(o_), n);
    };

    // ---- weight / input conversions ----
    cvt(adjf,  off_adjh, 1024 * 1024);
    cvt(Hf,    off_Hh,   1024 * 32);
    cvt(m_w,   off_mwh,  32 * 1024);
    cvt(gc1_w, off_gc1h, 32 * 256);
    cvt(gc2_w, off_gc2h, 256 * 32);
    cvt(lin_w, off_linh, 32 * 64);
    cvt(wih0,  off_wih0, 64 * 192);
    cvt(whh0,  off_whh0, 64 * 192);
    cvt(wih1,  off_wih1, 64 * 192);
    cvt(whh1,  off_whh1, 64 * 192);
    cvt(wq,    off_wqh,  64 * 64);
    cvt(wk,    off_wkh,  64 * 64);
    cvt(wv,    off_wvh,  64 * 64);
    cvt(fc_w,  off_fch,  64 * 64);
    cvt(l_w,   off_lwh,  1024 * 256);
    k_transpose_src<<<dim3(16384), dim3(256), 0, stream>>>(src, F16(off_xh));

    // ---- H_new = H @ m_w + m_b   [1024,1024] ----
    gemmL<true, false, false, true>(stream, F16(off_Hh), F16(off_mwh), F16(off_Hnewh),
                                    m_b, 1024, 1024, 32, 32, 1024, 1024, 0, 0, 0, 1);

    // ---- s1 = x @ gc1_w   [131072,256] (shared by both gcn passes) ----
    gemmL<false, false, false, true>(stream, F16(off_xh), F16(off_gc1h), F16(off_s1),
                                     nullptr, (int)NTOK, 256, 32, 32, 256, 256, 0, 0, 0, 1);

    // ---- two gcn passes: z (+)= A @ (relu(A@s1+b1) @ gc2_w) + b2 ----
    for (int pass = 0; pass < 2; ++pass) {
        const _Float16* Aadj = (pass == 0) ? F16(off_adjh) : F16(off_Hnewh);
        // h = relu(A @ s1 + gc1_b), batched over (b,t)
        gemmL<true, true, false, true>(stream, Aadj, F16(off_s1), F16(off_h),
                                       gc1_b, 1024, 256, 1024, 1024, 256, 256,
                                       0, 1024L * 256, 1024L * 256, 128);
        // s2 = h @ gc2_w
        gemmL<false, false, false, true>(stream, F16(off_h), F16(off_gc2h), F16(off_s2),
                                         nullptr, (int)NTOK, 32, 256, 256, 32, 32, 0, 0, 0, 1);
        // z (+)= A @ s2 + gc2_b, batched
        if (pass == 0)
            gemmL<true, false, false, false>(stream, Aadj, F16(off_s2), F32(off_z),
                                             gc2_b, 1024, 32, 1024, 1024, 32, 32,
                                             0, 1024L * 32, 1024L * 32, 128);
        else
            gemmL<true, false, true, false>(stream, Aadj, F16(off_s2), F32(off_z),
                                            gc2_b, 1024, 32, 1024, 1024, 32, 32,
                                            0, 1024L * 32, 1024L * 32, 128);
    }

    // ---- xlin = z @ lin_w + lin_b   [131072,64] ----
    k_f32_to_f16<<<dim3((NTOK * 32 + 255) / 256), dim3(256), 0, stream>>>(F32(off_z), F16(off_zh), (int)(NTOK * 32));
    gemmL<true, false, false, true>(stream, F16(off_zh), F16(off_linh), F16(off_xlin),
                                    lin_b, (int)NTOK, 64, 32, 32, 64, 64, 0, 0, 0, 1);

    // ---- 2-layer GRU over T=16 steps ----
    const size_t xin_off[2]  = { off_xlin, off_r1 };
    const size_t rout_off[2] = { off_r1,   off_r2 };
    const size_t wih_off[2]  = { off_wih0, off_wih1 };
    const size_t whh_off[2]  = { off_whh0, off_whh1 };
    const float* bih[2] = { bih0, bih1 };
    const float* bhh[2] = { bhh0, bhh1 };
    for (int layer = 0; layer < 2; ++layer) {
        k_zero_f16<<<dim3(2048), dim3(256), 0, stream>>>(F16(off_hst), 8192 * 64);
        for (int t = 0; t < 16; ++t) {
            // gi = x_t @ W_ih + b_ih  (batched over b: rows contiguous per (b,t))
            gemmL<true, false, false, false>(stream, F16(xin_off[layer]) + (long)t * 1024 * 64,
                                             F16(wih_off[layer]), F32(off_gi), bih[layer],
                                             1024, 192, 64, 64, 192, 192,
                                             16L * 1024 * 64, 0, 1024L * 192, 8);
            // gh = h @ W_hh + b_hh
            gemmL<true, false, false, false>(stream, F16(off_hst), F16(whh_off[layer]),
                                             F32(off_gh), bhh[layer],
                                             8192, 192, 64, 64, 192, 192, 0, 0, 0, 1);
            k_gru_gate<<<dim3(2048), dim3(256), 0, stream>>>(F32(off_gi), F32(off_gh),
                                                             F16(off_hst), F16(rout_off[layer]), t);
        }
    }

    // ---- temporal MHA ----
    gemmL<false, false, false, true>(stream, F16(off_r2), F16(off_wqh), F16(off_q),
                                     nullptr, (int)NTOK, 64, 64, 64, 64, 64, 0, 0, 0, 1);
    gemmL<false, false, false, true>(stream, F16(off_r2), F16(off_wkh), F16(off_k),
                                     nullptr, (int)NTOK, 64, 64, 64, 64, 64, 0, 0, 0, 1);
    gemmL<false, false, false, true>(stream, F16(off_r2), F16(off_wvh), F16(off_v),
                                     nullptr, (int)NTOK, 64, 64, 64, 64, 64, 0, 0, 0, 1);
    k_attention<<<dim3(2048), dim3(256), 0, stream>>>(F16(off_q), F16(off_k), F16(off_v), F16(off_o1));

    // o = attn_out @ fc_w + fc_b  (f32), then +residual, LN, write [B,S,T,64]
    gemmL<true, false, false, false>(stream, F16(off_o1), F16(off_fch), F32(off_ob),
                                     fc_b, (int)NTOK, 64, 64, 64, 64, 64, 0, 0, 0, 1);
    k_ln<<<dim3(512), dim3(256), 0, stream>>>(F32(off_ob), F16(off_r2), ln_g, ln_b, F16(off_olnb));

    // ---- hid = flat @ l_w + l_b   [8192,256] ----
    gemmL<true, false, false, true>(stream, F16(off_olnb), F16(off_lwh), F16(off_hid),
                                    l_b, 8192, 256, 1024, 1024, 256, 256, 0, 0, 0, 1);

    // ---- out = hid @ prj_w   [8192,2] ----
    k_proj<<<dim3(64), dim3(256), 0, stream>>>(F16(off_hid), prj_w, out);
}
